// GraphAttention_63548336111903
// MI455X (gfx1250) — matrix operand order repaired and checked by reading, not compiled, not run on hardware
//
#include <hip/hip_runtime.h>
#include <hip/hip_bf16.h>

// ---------------------------------------------------------------------------
// Graph attention block for MI455X (gfx1250, wave32).
// All GEMM-shaped math uses v_wmma_f32_16x16x32_bf16 (bf16 in, f32 accum).
// Segment softmax done with 3 edge passes + float atomics (fmax / fadd),
// recomputing the cheap 32x32 per-head logits from stored bf16 q/k so the
// 256MB logits tensor is never materialized.  LDS staging copies use the
// CDNA5 async global->LDS path (GLOBAL_LOAD_ASYNC_TO_LDS_B128 / ASYNCcnt).
// ---------------------------------------------------------------------------

typedef __bf16 bf16;
typedef __attribute__((ext_vector_type(16))) __bf16 v16bf;
typedef __attribute__((ext_vector_type(8)))  float  v8f;
typedef __attribute__((ext_vector_type(4)))  int    v4i;

#define DEVINL __device__ __forceinline__

static constexpr int Tt = 32;    // timesteps
static constexpr int Dd = 256;   // model dim
static constexpr int Hh = 8;     // heads
static constexpr int Kk = 32;    // head dim

#if defined(__gfx1250__) && __has_builtin(__builtin_amdgcn_global_load_async_to_lds_b128)
#define HAS_ASYNC_LDS 1
#else
#define HAS_ASYNC_LDS 0
#endif

DEVINL int lane_id() { return (int)(threadIdx.x & 31u); }
DEVINL int wave_id() { return (int)(threadIdx.x >> 5u); }

// 16-byte global -> LDS copy; async DMA on gfx1250 if available.
DEVINL void copy16(void* lds_dst, const void* gsrc) {
#if HAS_ASYNC_LDS
  __builtin_amdgcn_global_load_async_to_lds_b128(
      (__attribute__((address_space(1))) v4i*)(v4i*)const_cast<void*>(gsrc),
      (__attribute__((address_space(3))) v4i*)(v4i*)lds_dst, 0, 0);
#else
  *(uint4*)lds_dst = *(const uint4*)gsrc;
#endif
}

DEVINL void copy_join() {
#if HAS_ASYNC_LDS
#if __has_builtin(__builtin_amdgcn_s_wait_asynccnt)
  __builtin_amdgcn_s_wait_asynccnt(0);
#else
  asm volatile("s_wait_asynccnt 0x0" ::: "memory");
#endif
#endif
}

// A fragment: 16x32 bf16, row-major source, row stride ld.
// Layout: lane<16 -> M=lane, K 0..15; lane>=16 -> K 16..31 (matches B fragments).
DEVINL v16bf load_frag_a(const bf16* base, int ld) {
  int l = lane_id();
  const bf16* p = base + (size_t)(l & 15) * ld + ((l >> 4) << 4);
  v16bf a;
#pragma unroll
  for (int e = 0; e < 8; ++e) a[e] = p[e];
#pragma unroll
  for (int e = 0; e < 8; ++e) a[8 + e] = p[8 + e];
  return a;
}

// B fragment: 32x16 bf16 where source is stored "transposed" (row n = output
// column n, 32 contiguous k values).  lane<16 -> K 0..15, lane>=16 -> K 16..31.
DEVINL v16bf load_frag_b(const bf16* base, int ld) {
  int l = lane_id();
  const bf16* p = base + (size_t)(l & 15) * ld + ((l >> 4) << 4);
  v16bf b;
#pragma unroll
  for (int e = 0; e < 16; ++e) b[e] = p[e];
  return b;
}

// B fragment from a row-major [K x N] matrix (column per lane, strided reads).
DEVINL v16bf load_frag_b_strided(const bf16* base, int ld) {
  int l = lane_id();
  int n = l & 15, kb = (l >> 4) << 4;
  v16bf b;
#pragma unroll
  for (int e = 0; e < 16; ++e) b[e] = base[(size_t)(kb + e) * ld + n];
  return b;
}

DEVINL v8f wmma_bf16(v16bf a, v16bf b, v8f c) {
  return __builtin_amdgcn_wmma_f32_16x16x32_bf16(false, a, false, b, (short)0, c,
                                                 false, false);
}

DEVINL float wave_sum(float x) {
#pragma unroll
  for (int off = 16; off > 0; off >>= 1) x += __shfl_xor(x, off, 32);
  return x;
}

DEVINL float gelu_tanh(float x) {
  float x3 = x * x * x;
  return 0.5f * x * (1.0f + tanhf(0.79788456080286536f * (x + 0.044715f * x3)));
}

// ---------------------------------------------------------------------------
// K0: fp32 -> bf16 transpose (weights stored as [out, in] for contiguous B frags)
// ---------------------------------------------------------------------------
__global__ __launch_bounds__(256) void transpose_bf16(const float* __restrict__ src,
                                                      bf16* __restrict__ dst,
                                                      int rows, int cols) {
  int idx = blockIdx.x * 256 + threadIdx.x;
  if (idx >= rows * cols) return;
  int r = idx / cols, c = idx - r * cols;
  dst[(size_t)c * rows + r] = (bf16)src[idx];
}

__global__ __launch_bounds__(256) void fill_f32(float* __restrict__ p, float v, int n) {
  int i = blockIdx.x * 256 + threadIdx.x;
  if (i < n) p[i] = v;
}

// ---------------------------------------------------------------------------
// K1: per-row LayerNorm -> bf16   (one wave per 256-wide row)
// ---------------------------------------------------------------------------
__global__ __launch_bounds__(256) void ln_rows_bf16(const float* __restrict__ x,
                                                    const float* __restrict__ sc,
                                                    const float* __restrict__ of,
                                                    bf16* __restrict__ y, int nrows) {
  int row = blockIdx.x * 8 + wave_id();
  if (row >= nrows) return;
  int l = lane_id();
  const float* xr = x + (size_t)row * Dd;
  float v[8], s = 0.f, s2 = 0.f;
#pragma unroll
  for (int e = 0; e < 8; ++e) {
    float t = xr[l + 32 * e];
    v[e] = t; s += t; s2 += t * t;
  }
  s = wave_sum(s); s2 = wave_sum(s2);
  float mean = s * (1.f / Dd);
  float var  = s2 * (1.f / Dd) - mean * mean;
  float rstd = rsqrtf(var + 1e-5f);
  bf16* yr = y + (size_t)row * Dd;
#pragma unroll
  for (int e = 0; e < 8; ++e) {
    int c = l + 32 * e;
    yr[c] = (bf16)(sc[c] * (v[e] - mean) * rstd + of[c]);
  }
}

// ---------------------------------------------------------------------------
// K2: per-edge Q/K/V projections + logits + atomic segment-max.
// One workgroup (8 waves) per edge; B fragments shared across both row tiles.
// ---------------------------------------------------------------------------
__global__ __launch_bounds__(256) void edge_qkv_logits(
    const bf16* __restrict__ nodes_ln, const int* __restrict__ senders,
    const int* __restrict__ receivers, const bf16* __restrict__ WqT,
    const float* __restrict__ bq, const bf16* __restrict__ WkT,
    const float* __restrict__ bk, const bf16* __restrict__ WvT,
    const float* __restrict__ bv, const float* __restrict__ alibi,
    bf16* __restrict__ gq, bf16* __restrict__ gk, bf16* __restrict__ gv,
    float* __restrict__ segmax) {
  __shared__ __align__(16) bf16 s_sent[Tt * Dd];
  __shared__ __align__(16) bf16 s_recv[Tt * Dd];
  __shared__ __align__(16) bf16 s_q[Tt * Dd];
  __shared__ __align__(16) bf16 s_k[Tt * Dd];

  int e = blockIdx.x;
  int snd = senders[e], rcv = receivers[e];
  {  // stage the two gathered node tiles into LDS (async DMA path on CDNA5)
    const bf16* ps = nodes_ln + (size_t)snd * (Tt * Dd);
    const bf16* pr = nodes_ln + (size_t)rcv * (Tt * Dd);
    for (int i = threadIdx.x; i < Tt * Dd / 8; i += 256) {
      copy16(s_sent + i * 8, ps + i * 8);
      copy16(s_recv + i * 8, pr + i * 8);
    }
    copy_join();
  }
  __syncthreads();

  int w = wave_id(), l = lane_id();
  int n = l & 15, mo = (l >> 4) << 3;
  size_t gof = (size_t)e * (Tt * Dd);

  // ---- Q and K : [32 x 512] @ WT[256 x 512] -> [32 x 256]
  for (int qk = 0; qk < 2; ++qk) {
    const bf16* WT = qk ? WkT : WqT;
    const float* bias = qk ? bk : bq;
    bf16* sdst = qk ? s_k : s_q;
    bf16* gdst = qk ? gk : gq;
    for (int nt = w; nt < 16; nt += 8) {
      __builtin_prefetch(WT + (size_t)(nt * 16) * 512, 0, 0);
      v8f c0, c1;
      float bb = bias[nt * 16 + n];
#pragma unroll
      for (int i = 0; i < 8; ++i) { c0[i] = bb; c1[i] = bb; }
#pragma unroll
      for (int ks = 0; ks < 16; ++ks) {
        const bf16* asrc = (ks < 8) ? (s_sent + ks * 32)
                                    : (s_recv + (ks - 8) * 32);
        v16bf b  = load_frag_b(WT + (size_t)(nt * 16) * 512 + ks * 32, 512);
        v16bf a0 = load_frag_a(asrc, Dd);
        v16bf a1 = load_frag_a(asrc + 16 * Dd, Dd);
        c0 = wmma_bf16(a0, b, c0);
        c1 = wmma_bf16(a1, b, c1);
      }
#pragma unroll
      for (int i = 0; i < 8; ++i) {
        int col = nt * 16 + n;
        bf16 h0 = (bf16)c0[i], h1 = (bf16)c1[i];
        sdst[(mo + i) * Dd + col] = h0;
        sdst[(16 + mo + i) * Dd + col] = h1;
        gdst[gof + (mo + i) * Dd + col] = h0;
        gdst[gof + (16 + mo + i) * Dd + col] = h1;
      }
    }
  }

  // ---- V : [32 x 256] @ WvT[256 x 256] -> [32 x 256] (sent only)
  for (int nt = w; nt < 16; nt += 8) {
    __builtin_prefetch(WvT + (size_t)(nt * 16) * Dd, 0, 0);
    v8f c0, c1;
    float bb = bv[nt * 16 + n];
#pragma unroll
    for (int i = 0; i < 8; ++i) { c0[i] = bb; c1[i] = bb; }
#pragma unroll
    for (int ks = 0; ks < 8; ++ks) {
      v16bf b  = load_frag_b(WvT + (size_t)(nt * 16) * Dd + ks * 32, Dd);
      v16bf a0 = load_frag_a(s_sent + ks * 32, Dd);
      v16bf a1 = load_frag_a(s_sent + 16 * Dd + ks * 32, Dd);
      c0 = wmma_bf16(a0, b, c0);
      c1 = wmma_bf16(a1, b, c1);
    }
#pragma unroll
    for (int i = 0; i < 8; ++i) {
      int col = nt * 16 + n;
      gv[gof + (mo + i) * Dd + col] = (bf16)c0[i];
      gv[gof + (16 + mo + i) * Dd + col] = (bf16)c1[i];
    }
  }
  __syncthreads();

  // ---- logits[h,t,T'] = q_h . k_h^T ; one head per wave, 4 wmma from 2A+2B
  {
    int h = w;
    v16bf a[2], b[2];
#pragma unroll
    for (int mt = 0; mt < 2; ++mt) a[mt] = load_frag_a(s_q + mt * 16 * Dd + h * Kk, Dd);
#pragma unroll
    for (int nt = 0; nt < 2; ++nt) b[nt] = load_frag_b(s_k + nt * 16 * Dd + h * Kk, Dd);
#pragma unroll
    for (int mt = 0; mt < 2; ++mt) {
#pragma unroll
      for (int nt = 0; nt < 2; ++nt) {
        v8f c;
#pragma unroll
        for (int i = 0; i < 8; ++i) c[i] = 0.f;
        c = wmma_bf16(a[mt], b[nt], c);
#pragma unroll
        for (int i = 0; i < 8; ++i) {
          int t = mt * 16 + mo + i, Tp = nt * 16 + n;
          float lg = c[i] * 0.17677669529663687f + alibi[(h * Tt + t) * Tt + Tp];
          if (Tp > t) lg = -1e30f;
          size_t idx = (size_t)rcv * (Hh * Tt * Tt) + (size_t)h * (Tt * Tt) + t * Tt + Tp;
          __hip_atomic_fetch_max(&segmax[idx], lg, __ATOMIC_RELAXED,
                                 __HIP_MEMORY_SCOPE_AGENT);
        }
      }
    }
  }
}

// ---------------------------------------------------------------------------
// K3: recompute logits; atomic-add exp(l - segmax) into denom.
// ---------------------------------------------------------------------------
__global__ __launch_bounds__(256) void edge_denom(
    const bf16* __restrict__ gq, const bf16* __restrict__ gk,
    const int* __restrict__ receivers, const float* __restrict__ alibi,
    const float* __restrict__ segmax, float* __restrict__ denom) {
  int e = blockIdx.x, rcv = receivers[e];
  size_t gof = (size_t)e * (Tt * Dd);
  int w = wave_id(), l = lane_id();
  int n = l & 15, mo = (l >> 4) << 3;
  int h = w;
  v16bf a[2], b[2];
#pragma unroll
  for (int mt = 0; mt < 2; ++mt) a[mt] = load_frag_a(gq + gof + mt * 16 * Dd + h * Kk, Dd);
#pragma unroll
  for (int nt = 0; nt < 2; ++nt) b[nt] = load_frag_b(gk + gof + nt * 16 * Dd + h * Kk, Dd);
#pragma unroll
  for (int mt = 0; mt < 2; ++mt) {
#pragma unroll
    for (int nt = 0; nt < 2; ++nt) {
      v8f c;
#pragma unroll
      for (int i = 0; i < 8; ++i) c[i] = 0.f;
      c = wmma_bf16(a[mt], b[nt], c);
#pragma unroll
      for (int i = 0; i < 8; ++i) {
        int t = mt * 16 + mo + i, Tp = nt * 16 + n;
        float lg = c[i] * 0.17677669529663687f + alibi[(h * Tt + t) * Tt + Tp];
        if (Tp > t) lg = -1e30f;
        size_t idx = (size_t)rcv * (Hh * Tt * Tt) + (size_t)h * (Tt * Tt) + t * Tt + Tp;
        atomicAdd(&denom[idx], __expf(lg - segmax[idx]));
      }
    }
  }
}

// ---------------------------------------------------------------------------
// K4: recompute logits -> attn weights; messages = attn @ v; atomic scatter.
// ---------------------------------------------------------------------------
__global__ __launch_bounds__(256) void edge_messages(
    const bf16* __restrict__ gq, const bf16* __restrict__ gk,
    const bf16* __restrict__ gv, const int* __restrict__ receivers,
    const float* __restrict__ alibi, const float* __restrict__ segmax,
    const float* __restrict__ denom, float* __restrict__ nodes_attn) {
  __shared__ __align__(16) bf16 s_attn[Hh * Tt * Tt];
  __shared__ __align__(16) bf16 s_v[Tt * Dd];

  int e = blockIdx.x, rcv = receivers[e];
  size_t gof = (size_t)e * (Tt * Dd);
  {
    const bf16* pv = gv + gof;
    for (int i = threadIdx.x; i < Tt * Dd / 8; i += 256)
      copy16(s_v + i * 8, pv + i * 8);
    copy_join();
  }
  int w = wave_id(), l = lane_id();
  int n = l & 15, mo = (l >> 4) << 3;
  int h = w;

  {  // attention weights for this head
    v16bf a[2], b[2];
#pragma unroll
    for (int mt = 0; mt < 2; ++mt) a[mt] = load_frag_a(gq + gof + mt * 16 * Dd + h * Kk, Dd);
#pragma unroll
    for (int nt = 0; nt < 2; ++nt) b[nt] = load_frag_b(gk + gof + nt * 16 * Dd + h * Kk, Dd);
#pragma unroll
    for (int mt = 0; mt < 2; ++mt) {
#pragma unroll
      for (int nt = 0; nt < 2; ++nt) {
        v8f c;
#pragma unroll
        for (int i = 0; i < 8; ++i) c[i] = 0.f;
        c = wmma_bf16(a[mt], b[nt], c);
#pragma unroll
        for (int i = 0; i < 8; ++i) {
          int t = mt * 16 + mo + i, Tp = nt * 16 + n;
          float lg = c[i] * 0.17677669529663687f + alibi[(h * Tt + t) * Tt + Tp];
          if (Tp > t) lg = -1e30f;
          size_t idx = (size_t)rcv * (Hh * Tt * Tt) + (size_t)h * (Tt * Tt) + t * Tt + Tp;
          float attn = __expf(lg - segmax[idx]) / denom[idx];
          s_attn[h * (Tt * Tt) + t * Tt + Tp] = (bf16)attn;
        }
      }
    }
  }
  __syncthreads();

  // messages[t, h*32 + d] = sum_T attn[h,t,T] * v[T, h*32 + d]
  {
    v16bf a[2], b[2];
#pragma unroll
    for (int mt = 0; mt < 2; ++mt)
      a[mt] = load_frag_a(s_attn + h * (Tt * Tt) + mt * 16 * Tt, Tt);
#pragma unroll
    for (int n2 = 0; n2 < 2; ++n2)
      b[n2] = load_frag_b_strided(s_v + h * Kk + n2 * 16, Dd);
#pragma unroll
    for (int mt = 0; mt < 2; ++mt) {
#pragma unroll
      for (int n2 = 0; n2 < 2; ++n2) {
        v8f c;
#pragma unroll
        for (int i = 0; i < 8; ++i) c[i] = 0.f;
        c = wmma_bf16(a[mt], b[n2], c);
#pragma unroll
        for (int i = 0; i < 8; ++i) {
          size_t idx = (size_t)rcv * (Tt * Dd) + (size_t)(mt * 16 + mo + i) * Dd +
                       h * Kk + n2 * 16 + n;
          atomicAdd(&nodes_attn[idx], c[i]);
        }
      }
    }
  }
}

// ---------------------------------------------------------------------------
// K5: fused residual + LN + MLP (gelu) + residual + final LN.  16 rows/block.
// ---------------------------------------------------------------------------
__global__ __launch_bounds__(256) void mlp_out(
    const float* __restrict__ nodes, const float* __restrict__ nodes_attn,
    const float* __restrict__ lnm_s, const float* __restrict__ lnm_o,
    const bf16* __restrict__ W1T, const float* __restrict__ b1,
    const bf16* __restrict__ W2T, const float* __restrict__ b2,
    const float* __restrict__ lno_s, const float* __restrict__ lno_o,
    float* __restrict__ out) {
  __shared__ __align__(16) bf16 s_x[16 * Dd];       // LN(nodes_attn) bf16
  __shared__ __align__(16) float s_res[16 * Dd];    // nodes + attn (+ dense)
  __shared__ __align__(16) bf16 s_h[16 * 1024];     // gelu hidden bf16

  size_t row0 = (size_t)blockIdx.x * 16;
  int w = wave_id(), l = lane_id();
  int n = l & 15, mo = (l >> 4) << 3;

  for (int rr = w; rr < 16; rr += 8) {
    const float* xa = nodes_attn + (row0 + rr) * Dd;
    const float* xn = nodes + (row0 + rr) * Dd;
    float v[8], s = 0.f, s2 = 0.f;
#pragma unroll
    for (int e = 0; e < 8; ++e) {
      int c = l + 32 * e;
      float t = xa[c];
      v[e] = t; s += t; s2 += t * t;
      s_res[rr * Dd + c] = xn[c] + t;
    }
    s = wave_sum(s); s2 = wave_sum(s2);
    float mean = s * (1.f / Dd);
    float rstd = rsqrtf(s2 * (1.f / Dd) - mean * mean + 1e-5f);
#pragma unroll
    for (int e = 0; e < 8; ++e) {
      int c = l + 32 * e;
      s_x[rr * Dd + c] = (bf16)(lnm_s[c] * (v[e] - mean) * rstd + lnm_o[c]);
    }
  }
  __syncthreads();

  // GEMM1: [16 x 256] @ W1T[1024 x 256] -> gelu -> s_h [16 x 1024]
  {
    v16bf a[8];
#pragma unroll
    for (int ks = 0; ks < 8; ++ks) a[ks] = load_frag_a(s_x + ks * 32, Dd);
    for (int nt = w; nt < 64; nt += 8) {
      __builtin_prefetch(W1T + (size_t)(nt * 16) * Dd, 0, 0);
      v8f c;
      float bb = b1[nt * 16 + n];
#pragma unroll
      for (int i = 0; i < 8; ++i) c[i] = bb;
#pragma unroll
      for (int ks = 0; ks < 8; ++ks) {
        v16bf b = load_frag_b(W1T + (size_t)(nt * 16) * Dd + ks * 32, Dd);
        c = wmma_bf16(a[ks], b, c);
      }
#pragma unroll
      for (int i = 0; i < 8; ++i)
        s_h[(mo + i) * 1024 + nt * 16 + n] = (bf16)gelu_tanh(c[i]);
    }
  }
  __syncthreads();

  // GEMM2: [16 x 1024] @ W2T[256 x 1024] -> += s_res  (A shared across 2 tiles)
  {
    int nt0 = w, nt1 = w + 8;
    v8f c0, c1;
    float b0 = b2[nt0 * 16 + n], b1v = b2[nt1 * 16 + n];
#pragma unroll
    for (int i = 0; i < 8; ++i) { c0[i] = b0; c1[i] = b1v; }
#pragma unroll
    for (int ks = 0; ks < 32; ++ks) {
      v16bf a  = load_frag_a(s_h + ks * 32, 1024);
      v16bf bb0 = load_frag_b(W2T + (size_t)(nt0 * 16) * 1024 + ks * 32, 1024);
      v16bf bb1 = load_frag_b(W2T + (size_t)(nt1 * 16) * 1024 + ks * 32, 1024);
      c0 = wmma_bf16(a, bb0, c0);
      c1 = wmma_bf16(a, bb1, c1);
    }
#pragma unroll
    for (int i = 0; i < 8; ++i) {
      s_res[(mo + i) * Dd + nt0 * 16 + n] += c0[i];
      s_res[(mo + i) * Dd + nt1 * 16 + n] += c1[i];
    }
  }
  __syncthreads();

  // final LN -> output
  for (int rr = w; rr < 16; rr += 8) {
    float v[8], s = 0.f, s2 = 0.f;
#pragma unroll
    for (int e = 0; e < 8; ++e) {
      float t = s_res[rr * Dd + l + 32 * e];
      v[e] = t; s += t; s2 += t * t;
    }
    s = wave_sum(s); s2 = wave_sum(s2);
    float mean = s * (1.f / Dd);
    float rstd = rsqrtf(s2 * (1.f / Dd) - mean * mean + 1e-5f);
    float* orow = out + (row0 + rr) * Dd;
#pragma unroll
    for (int e = 0; e < 8; ++e) {
      int c = l + 32 * e;
      orow[c] = lno_s[c] * (v[e] - mean) * rstd + lno_o[c];
    }
  }
}

// ---------------------------------------------------------------------------
extern "C" void kernel_launch(void* const* d_in, const int* in_sizes, int n_in,
                              void* d_out, int out_size, void* d_ws, size_t ws_size,
                              hipStream_t stream) {
  const float* nodes   = (const float*)d_in[0];
  const int* senders   = (const int*)d_in[1];
  const int* receivers = (const int*)d_in[2];
  const float* alibi   = (const float*)d_in[3];
  const float* lnq_s   = (const float*)d_in[4];
  const float* lnq_o   = (const float*)d_in[5];
  const float* Wq      = (const float*)d_in[6];
  const float* bq      = (const float*)d_in[7];
  const float* Wk      = (const float*)d_in[8];
  const float* bk      = (const float*)d_in[9];
  const float* Wv      = (const float*)d_in[10];
  const float* bv      = (const float*)d_in[11];
  const float* lnm_s   = (const float*)d_in[12];
  const float* lnm_o   = (const float*)d_in[13];
  const float* W1      = (const float*)d_in[14];
  const float* b1      = (const float*)d_in[15];
  const float* W2      = (const float*)d_in[16];
  const float* b2      = (const float*)d_in[17];
  const float* lno_s   = (const float*)d_in[18];
  const float* lno_o   = (const float*)d_in[19];
  float* out = (float*)d_out;

  const int E  = in_sizes[1];                 // 8192
  const int NT = in_sizes[0] / Dd;            // 65536 rows
  const int Nn = NT / Tt;                     // 2048 nodes

  char* ws = (char*)d_ws;
  size_t off = 0;
  auto alloc = [&](size_t bytes) -> void* {
    void* p = ws + off;
    off = (off + bytes + 255) & ~(size_t)255;
    return p;
  };
  bf16* WqT = (bf16*)alloc((size_t)512 * 256 * 2);
  bf16* WkT = (bf16*)alloc((size_t)512 * 256 * 2);
  bf16* WvT = (bf16*)alloc((size_t)256 * 256 * 2);
  bf16* W1T = (bf16*)alloc((size_t)256 * 1024 * 2);
  bf16* W2T = (bf16*)alloc((size_t)1024 * 256 * 2);
  bf16* nodes_ln = (bf16*)alloc((size_t)NT * Dd * 2);
  bf16* gq = (bf16*)alloc((size_t)E * Tt * Dd * 2);
  bf16* gk = (bf16*)alloc((size_t)E * Tt * Dd * 2);
  bf16* gv = (bf16*)alloc((size_t)E * Tt * Dd * 2);
  float* segmax = (float*)alloc((size_t)Nn * Hh * Tt * Tt * 4);
  float* denom  = (float*)alloc((size_t)Nn * Hh * Tt * Tt * 4);
  float* nodes_attn = (float*)alloc((size_t)NT * Dd * 4);

  // weights -> transposed bf16
  transpose_bf16<<<(512 * 256 + 255) / 256, 256, 0, stream>>>(Wq, WqT, 512, 256);
  transpose_bf16<<<(512 * 256 + 255) / 256, 256, 0, stream>>>(Wk, WkT, 512, 256);
  transpose_bf16<<<(256 * 256 + 255) / 256, 256, 0, stream>>>(Wv, WvT, 256, 256);
  transpose_bf16<<<(256 * 1024 + 255) / 256, 256, 0, stream>>>(W1, W1T, 256, 1024);
  transpose_bf16<<<(1024 * 256 + 255) / 256, 256, 0, stream>>>(W2, W2T, 1024, 256);

  // LN(nodes) -> bf16
  ln_rows_bf16<<<(NT + 7) / 8, 256, 0, stream>>>(nodes, lnq_s, lnq_o, nodes_ln, NT);

  // init segment buffers
  const int segN = Nn * Hh * Tt * Tt;
  (void)hipMemsetAsync(denom, 0, (size_t)segN * 4, stream);
  (void)hipMemsetAsync(nodes_attn, 0, (size_t)NT * Dd * 4, stream);
  fill_f32<<<(segN + 255) / 256, 256, 0, stream>>>(segmax, -__builtin_huge_valf(), segN);

  // pass A: QKV + logits + seg-max
  edge_qkv_logits<<<E, 256, 0, stream>>>(nodes_ln, senders, receivers, WqT, bq, WkT,
                                         bk, WvT, bv, alibi, gq, gk, gv, segmax);
  // pass B: denominators
  edge_denom<<<E, 256, 0, stream>>>(gq, gk, receivers, alibi, segmax, denom);
  // pass C: attn @ v scattered into nodes_attn
  edge_messages<<<E, 256, 0, stream>>>(gq, gk, gv, receivers, alibi, segmax, denom,
                                       nodes_attn);
  // MLP + residuals + final LN
  mlp_out<<<NT / 16, 256, 0, stream>>>(nodes, nodes_attn, lnm_s, lnm_o, W1T, b1,
                                       W2T, b2, lno_s, lno_o, out);
}
